// AdpHeadedAttention_38920993636354
// MI455X (gfx1250) — compile-verified, matrix operand order checked by reading, not run on hardware
//
#include <hip/hip_runtime.h>
#include <hip/hip_bf16.h>
#include <math.h>

#define B_  4
#define T_  1024
#define F_  1024
#define H_  16
#define DK_ 64

typedef __attribute__((ext_vector_type(16))) __bf16 v16bf;
typedef __attribute__((ext_vector_type(8)))  __bf16 v8bf;
typedef __attribute__((ext_vector_type(8)))  float  v8f;

// Optional CDNA5 async global->LDS path (ASYNCcnt-tracked), guarded so the
// file still compiles if the toolchain doesn't declare the builtins.
// Probe result (round 2): builtin exists, arg0 must be AS(1) pointer to
// int __attribute__((vector_size(16))).
#if defined(__has_builtin)
#if __has_builtin(__builtin_amdgcn_global_load_async_to_lds_b128) && \
    __has_builtin(__builtin_amdgcn_s_wait_asynccnt)
#define USE_ASYNC_LDS 1
#endif
#endif
#ifndef USE_ASYNC_LDS
#define USE_ASYNC_LDS 0
#endif

#if USE_ASYNC_LDS
typedef int v4i_vs __attribute__((__vector_size__(16)));
typedef __attribute__((address_space(1))) v4i_vs* gptr_b128;
typedef __attribute__((address_space(3))) v4i_vs* lptr_b128;
#endif

// ---------------- fragment helpers ----------------
// A/B 16-bit frag layout (ISA 7.12.2): lane l (lo=l&15, hi=l>>4) holds
// halfs j -> k = (j<8) ? hi*8+j : 16+hi*8+(j-8)  within a K=32 slab.
// C/D f32 frag: lane col = l&15, vgpr r -> row = r + 8*hi.

__device__ inline v8bf ld8(const __bf16* p) { return *(const v8bf*)p; }

__device__ inline v8bf cvt8(const float* p) {
  float4 x = *(const float4*)p;
  float4 y = *(const float4*)(p + 4);
  v8bf r;
  r[0] = (__bf16)x.x; r[1] = (__bf16)x.y; r[2] = (__bf16)x.z; r[3] = (__bf16)x.w;
  r[4] = (__bf16)y.x; r[5] = (__bf16)y.y; r[6] = (__bf16)y.z; r[7] = (__bf16)y.w;
  return r;
}

__device__ inline v16bf mk16(v8bf a, v8bf b) {
  v16bf r;
#pragma unroll
  for (int i = 0; i < 8; i++) { r[i] = a[i]; r[i + 8] = b[i]; }
  return r;
}

__device__ inline v8f wmma_bf16(v16bf a, v16bf b, v8f c) {
  return __builtin_amdgcn_wmma_f32_16x16x32_bf16(false, a, false, b, (short)0, c,
                                                 false, false);
}

__device__ inline float redmax16(float v) {
#pragma unroll
  for (int m = 1; m < 16; m <<= 1) v = fmaxf(v, __shfl_xor(v, m, 16));
  return v;
}
__device__ inline float redsum16(float v) {
#pragma unroll
  for (int m = 1; m < 16; m <<= 1) v += __shfl_xor(v, m, 16);
  return v;
}

// ---------------- kernel 0: fp32 -> bf16 weight conversion ----------------
// grid (F*F/(256*8), 4), block 256. One pass over Wq,Wk,Wv,Wo.
__global__ __launch_bounds__(256) void cvt_weights(
    const float* __restrict__ Wq, const float* __restrict__ Wk,
    const float* __restrict__ Wv, const float* __restrict__ Wo,
    unsigned short* Wb_) {
  const int sel = blockIdx.y;
  const float* src = sel == 0 ? Wq : (sel == 1 ? Wk : (sel == 2 ? Wv : Wo));
  __bf16* dst = (__bf16*)Wb_ + (size_t)sel * F_ * F_;
  const size_t idx = ((size_t)blockIdx.x * 256 + threadIdx.x) * 8;
  *(v8bf*)(dst + idx) = cvt8(src + idx);
}

// ---------------- kernel 1: norms / a_q mask / scales ----------------
__global__ __launch_bounds__(256) void prep_kernel(
    const float* __restrict__ Wq, const float* __restrict__ Wk,
    const float* __restrict__ a_q, float* __restrict__ qscale,
    float* __restrict__ kscale, float* __restrict__ out /* d_out base */) {
  __shared__ float red[256];
  const int tid = threadIdx.x;

  float nq[4], nk[4], aq[4];
  float amax = 0.0f;
#pragma unroll
  for (int i = 0; i < 4; i++) {
    const int f = tid + i * 256;
    const float4* wq4 = (const float4*)(Wq + (size_t)f * F_);
    const float4* wk4 = (const float4*)(Wk + (size_t)f * F_);
    float sq = 0.0f, sk = 0.0f;
    for (int c = 0; c < F_ / 4; c++) {
      float4 a = wq4[c];
      sq += a.x * a.x + a.y * a.y + a.z * a.z + a.w * a.w;
      float4 b = wk4[c];
      sk += b.x * b.x + b.y * b.y + b.z * b.z + b.w * b.w;
    }
    nq[i] = sqrtf(sq);
    nk[i] = sqrtf(sk);
    aq[i] = a_q[f];
    amax = fmaxf(amax, fabsf(aq[i]));
  }
  red[tid] = amax;
  __syncthreads();
  for (int s = 128; s > 0; s >>= 1) {
    if (tid < s) red[tid] = fmaxf(red[tid], red[tid + s]);
    __syncthreads();
  }
  amax = red[0];
  __syncthreads();

  const float thr = 0.001f * amax;
  float* lqk_out = out + (size_t)B_ * T_ * F_;
  float* aqf_out = lqk_out + 1;

  float l1 = 0.0f;
#pragma unroll
  for (int i = 0; i < 4; i++) {
    const int f = tid + i * 256;
    const float af = (fabsf(aq[i]) > thr) ? aq[i] : 0.0f;
    l1 += fabsf(af);
    aqf_out[f] = af;
    qscale[f] = af / nq[i] * 0.125f;  // fold 1/sqrt(DK) into q scale
    kscale[f] = 1.0f / nk[i];
  }
  red[tid] = l1;
  __syncthreads();
  for (int s = 128; s > 0; s >>= 1) {
    if (tid < s) red[tid] += red[tid + s];
    __syncthreads();
  }
  if (tid == 0) lqk_out[0] = red[0];
}

// ---------------- kernel 2: Q/K/V projections (bf16 WMMA) ----------------
// grid (tokens/128, F/64, 3), block 128 (4 waves). Wave -> 32x64 output tile
// (two A row-groups share each B fragment -> 8 WMMA per K-step).
__global__ __launch_bounds__(128) void proj_qkv(
    const float* __restrict__ q_in, const float* __restrict__ k_in,
    const float* __restrict__ v_in, const unsigned short* Wqb_,
    const unsigned short* Wkb_, const unsigned short* Wvb_,
    const float* __restrict__ bq, const float* __restrict__ bk,
    const float* __restrict__ bv, const float* __restrict__ qscale,
    const float* __restrict__ kscale, unsigned short* Qb_, unsigned short* Kb_,
    unsigned short* Vb_) {
  const int sel = blockIdx.z;
  const float* X = sel == 0 ? q_in : (sel == 1 ? k_in : v_in);
  const __bf16* W =
      (const __bf16*)(sel == 0 ? Wqb_ : (sel == 1 ? Wkb_ : Wvb_));
  const float* bias = sel == 0 ? bq : (sel == 1 ? bk : bv);
  const float* scale = sel == 0 ? qscale : (sel == 1 ? kscale : nullptr);
  __bf16* Out = (__bf16*)(sel == 0 ? Qb_ : (sel == 1 ? Kb_ : Vb_));

  const int lane = threadIdx.x & 31;
  const int wave = threadIdx.x >> 5;
  const int lo = lane & 15, hi = lane >> 4;
  const int m0 = blockIdx.x * 128 + wave * 32;  // token tile base (32 rows)
  const int n0 = blockIdx.y * 64;               // output feature tile base

  const float* arow0 = X + (size_t)(m0 + lo) * F_;
  const float* arow1 = X + (size_t)(m0 + 16 + lo) * F_;

  const v8f zero = {};
  v8f acc[2][4];
#pragma unroll
  for (int g = 0; g < 2; g++)
#pragma unroll
    for (int c = 0; c < 4; c++) acc[g][c] = zero;

  for (int kk = 0; kk < F_; kk += 32) {
    const int kb1 = kk + hi * 8;
    const int kb2 = kk + 16 + hi * 8;
    __builtin_prefetch(arow0 + kb1 + 64, 0, 1);
    const v16bf A0 = mk16(cvt8(arow0 + kb1), cvt8(arow0 + kb2));
    const v16bf A1 = mk16(cvt8(arow1 + kb1), cvt8(arow1 + kb2));
#pragma unroll
    for (int c = 0; c < 4; c++) {
      const __bf16* wrow = W + (size_t)(n0 + c * 16 + lo) * F_;
      const v16bf Bf = mk16(ld8(wrow + kb1), ld8(wrow + kb2));
      acc[0][c] = wmma_bf16(A0, Bf, acc[0][c]);
      acc[1][c] = wmma_bf16(A1, Bf, acc[1][c]);
    }
  }

#pragma unroll
  for (int c = 0; c < 4; c++) {
    const int n = n0 + c * 16 + lo;
    const float bn = bias[n];
    const float sn = scale ? scale[n] : 1.0f;
    const int h = n >> 6, d = n & 63;
#pragma unroll
    for (int g = 0; g < 2; g++) {
#pragma unroll
      for (int r = 0; r < 8; r++) {
        const int tg = m0 + g * 16 + hi * 8 + r;  // token in [0, B*T)
        const int b = tg >> 10, t = tg & (T_ - 1);
        const float v = (acc[g][c][r] + bn) * sn;
        Out[(((size_t)(b * H_ + h)) * T_ + t) * DK_ + d] = (__bf16)v;
      }
    }
  }
}

// ---------------- kernel 3: flash attention ----------------
// grid (T/64, B*H), block 128 (4 waves); wave handles 16 q rows.
__global__ __launch_bounds__(128) void attn_kernel(
    const unsigned short* Qb_, const unsigned short* Kb_,
    const unsigned short* Vb_, const int* __restrict__ mask,
    unsigned short* Xb_) {
  __shared__ __bf16 Kt[32 * 64];      // [key][feat]
  __shared__ __bf16 Vt[64 * 32];      // transposed: [feat][key]
  __shared__ __bf16 Pr[4][16 * 32];   // per-wave probs [row][key]

  const __bf16* Qb = (const __bf16*)Qb_;
  const __bf16* Kb = (const __bf16*)Kb_;
  const __bf16* Vb = (const __bf16*)Vb_;
  __bf16* Xb = (__bf16*)Xb_;

  const int bh = blockIdx.y;           // b*H + h
  const int b = bh >> 4;
  const int h = bh & 15;
  const int wave = threadIdx.x >> 5, lane = threadIdx.x & 31;
  const int lo = lane & 15, hi = lane >> 4;
  const int q0 = blockIdx.x * 64 + wave * 16;

  // Q fragments for this wave's 16 rows, K=64 -> two 16x32 slabs
  const __bf16* qrow = Qb + ((size_t)bh * T_ + (q0 + lo)) * DK_;
  const v16bf QA0 = mk16(ld8(qrow + hi * 8), ld8(qrow + 16 + hi * 8));
  const v16bf QA1 = mk16(ld8(qrow + 32 + hi * 8), ld8(qrow + 48 + hi * 8));

  const v8f zero = {};
  v8f acc[4];
#pragma unroll
  for (int c = 0; c < 4; c++) acc[c] = zero;
  float rowm[8], rowsum[8];
#pragma unroll
  for (int r = 0; r < 8; r++) { rowm[r] = -3.0e38f; rowsum[r] = 0.0f; }

  const int* mrow = mask + b * T_;

  for (int k0 = 0; k0 < T_; k0 += 32) {
    // cooperative load of K tile + transposed V tile
    {
      const int t = threadIdx.x;
      const int key = t >> 2;
      const int fq = (t & 3) * 16;
      const __bf16* ksrc = Kb + ((size_t)bh * T_ + (k0 + key)) * DK_ + fq;
      const __bf16* vsrc = Vb + ((size_t)bh * T_ + (k0 + key)) * DK_ + fq;
#if USE_ASYNC_LDS
      __builtin_amdgcn_global_load_async_to_lds_b128(
          (gptr_b128)ksrc, (lptr_b128)(Kt + key * 64 + fq), 0, 0);
      __builtin_amdgcn_global_load_async_to_lds_b128(
          (gptr_b128)(ksrc + 8), (lptr_b128)(Kt + key * 64 + fq + 8), 0, 0);
#else
      *(v8bf*)(Kt + key * 64 + fq) = *(const v8bf*)ksrc;
      *(v8bf*)(Kt + key * 64 + fq + 8) = *(const v8bf*)(ksrc + 8);
#endif
#pragma unroll
      for (int i = 0; i < 16; i++) Vt[(fq + i) * 32 + key] = vsrc[i];
#if USE_ASYNC_LDS
      __builtin_amdgcn_s_wait_asynccnt(0);
#endif
    }
    __syncthreads();

    // scores: two 16x16 key tiles, each accumulating over K=64 (2 WMMA)
    v8f s[2];
    float keep[2];
#pragma unroll
    for (int t = 0; t < 2; t++) {
      const int krel = t * 16 + lo;
      const __bf16* kr = Kt + krel * 64;
      const v16bf B0 = mk16(ld8(kr + hi * 8), ld8(kr + 16 + hi * 8));
      const v16bf B1 = mk16(ld8(kr + 32 + hi * 8), ld8(kr + 48 + hi * 8));
      v8f z = zero;
      z = wmma_bf16(QA0, B0, z);
      z = wmma_bf16(QA1, B1, z);
      const int mv = mrow[k0 + krel];
      keep[t] = mv ? 1.0f : 0.0f;
#pragma unroll
      for (int r = 0; r < 8; r++) s[t][r] = mv ? z[r] : -3.0e38f;
    }

    // online softmax: new row max, probs, row sums, accumulator rescale
    float newm[8], psum[8];
#pragma unroll
    for (int r = 0; r < 8; r++) {
      float mx = fmaxf(s[0][r], s[1][r]);
      mx = redmax16(mx);
      newm[r] = fmaxf(rowm[r], mx);
      psum[r] = 0.0f;
    }
#pragma unroll
    for (int t = 0; t < 2; t++) {
#pragma unroll
      for (int r = 0; r < 8; r++) {
        const float p = __expf(s[t][r] - newm[r]) * keep[t];
        psum[r] += p;
        Pr[wave][(hi * 8 + r) * 32 + t * 16 + lo] = (__bf16)p;
      }
    }
#pragma unroll
    for (int r = 0; r < 8; r++) {
      const float ps = redsum16(psum[r]);
      const float sc = __expf(rowm[r] - newm[r]);
      rowsum[r] = rowsum[r] * sc + ps;
      rowm[r] = newm[r];
#pragma unroll
      for (int c = 0; c < 4; c++) acc[c][r] *= sc;
    }

    // re-fragment probs (D layout -> A layout) via LDS, then attn @ V
    const __bf16* pr = Pr[wave] + lo * 32;
    const v16bf PA = mk16(ld8(pr + hi * 8), ld8(pr + 16 + hi * 8));
#pragma unroll
    for (int c = 0; c < 4; c++) {
      const __bf16* vr = Vt + (c * 16 + lo) * 32;
      const v16bf VB = mk16(ld8(vr + hi * 8), ld8(vr + 16 + hi * 8));
      acc[c] = wmma_bf16(PA, VB, acc[c]);
    }
    __syncthreads();
  }

  // finalize: divide by row sums, store bf16 x in [B,T,F] token-major layout
#pragma unroll
  for (int r = 0; r < 8; r++) {
    const float inv = rowsum[r] > 0.0f ? 1.0f / rowsum[r] : 0.0f;
    const int tg = q0 + hi * 8 + r;
#pragma unroll
    for (int c = 0; c < 4; c++) {
      const int f = h * 64 + c * 16 + lo;
      Xb[((size_t)b * T_ + tg) * F_ + f] = (__bf16)(acc[c][r] * inv);
    }
  }
}

// ---------------- kernel 4: output projection ----------------
// grid (tokens/128, F/64), block 128 (4 waves); wave -> 32x64 tile.
__global__ __launch_bounds__(128) void out_gemm(
    const unsigned short* Xb_, const unsigned short* Wob_,
    const float* __restrict__ bo, float* __restrict__ out) {
  const __bf16* Xb = (const __bf16*)Xb_;
  const __bf16* Wo = (const __bf16*)Wob_;

  const int lane = threadIdx.x & 31;
  const int wave = threadIdx.x >> 5;
  const int lo = lane & 15, hi = lane >> 4;
  const int m0 = blockIdx.x * 128 + wave * 32;
  const int n0 = blockIdx.y * 64;

  const __bf16* arow0 = Xb + (size_t)(m0 + lo) * F_;
  const __bf16* arow1 = Xb + (size_t)(m0 + 16 + lo) * F_;

  const v8f zero = {};
  v8f acc[2][4];
#pragma unroll
  for (int g = 0; g < 2; g++)
#pragma unroll
    for (int c = 0; c < 4; c++) acc[g][c] = zero;

  for (int kk = 0; kk < F_; kk += 32) {
    const int kb1 = kk + hi * 8;
    const int kb2 = kk + 16 + hi * 8;
    const v16bf A0 = mk16(ld8(arow0 + kb1), ld8(arow0 + kb2));
    const v16bf A1 = mk16(ld8(arow1 + kb1), ld8(arow1 + kb2));
#pragma unroll
    for (int c = 0; c < 4; c++) {
      const __bf16* wrow = Wo + (size_t)(n0 + c * 16 + lo) * F_;
      const v16bf Bf = mk16(ld8(wrow + kb1), ld8(wrow + kb2));
      acc[0][c] = wmma_bf16(A0, Bf, acc[0][c]);
      acc[1][c] = wmma_bf16(A1, Bf, acc[1][c]);
    }
  }

#pragma unroll
  for (int c = 0; c < 4; c++) {
    const int n = n0 + c * 16 + lo;
    const float bn = bo[n];
#pragma unroll
    for (int g = 0; g < 2; g++) {
#pragma unroll
      for (int r = 0; r < 8; r++) {
        const int tg = m0 + g * 16 + hi * 8 + r;
        out[(size_t)tg * F_ + n] = acc[g][c][r] + bn;
      }
    }
  }
}

// ---------------- launcher ----------------
extern "C" void kernel_launch(void* const* d_in, const int* in_sizes, int n_in,
                              void* d_out, int out_size, void* d_ws,
                              size_t ws_size, hipStream_t stream) {
  const float* query = (const float*)d_in[0];
  const float* key   = (const float*)d_in[1];
  const float* value = (const float*)d_in[2];
  const int*   mask  = (const int*)d_in[3];
  const float* Wq = (const float*)d_in[4];
  const float* bq = (const float*)d_in[5];
  const float* Wk = (const float*)d_in[6];
  const float* bk = (const float*)d_in[7];
  const float* Wv = (const float*)d_in[8];
  const float* bv = (const float*)d_in[9];
  const float* Wo = (const float*)d_in[10];
  const float* bo = (const float*)d_in[11];
  const float* a_q = (const float*)d_in[12];
  float* out = (float*)d_out;

  char* ws = (char*)d_ws;
  const size_t W_ELEMS = (size_t)F_ * F_;                // 1 Mi per weight
  const size_t QKV_ELEMS = (size_t)B_ * H_ * T_ * DK_;   // 4 Mi bf16 elements
  float* qscale = (float*)ws;                            // 4 KB
  float* kscale = qscale + F_;                           // 4 KB
  unsigned short* Wb = (unsigned short*)(ws + 8192);     // 4 x 1Mi bf16 = 8 MB
  unsigned short* Wqb = Wb;
  unsigned short* Wkb = Wb + W_ELEMS;
  unsigned short* Wvb = Wb + 2 * W_ELEMS;
  unsigned short* Wob = Wb + 3 * W_ELEMS;
  unsigned short* Qb = Wb + 4 * W_ELEMS;
  unsigned short* Kb = Qb + QKV_ELEMS;
  unsigned short* Vb = Kb + QKV_ELEMS;
  unsigned short* Xb = Vb + QKV_ELEMS;

  cvt_weights<<<dim3((F_ * F_) / (256 * 8), 4), 256, 0, stream>>>(Wq, Wk, Wv,
                                                                  Wo, Wb);

  prep_kernel<<<1, 256, 0, stream>>>(Wq, Wk, a_q, qscale, kscale, out);

  proj_qkv<<<dim3((B_ * T_) / 128, F_ / 64, 3), 128, 0, stream>>>(
      query, key, value, Wqb, Wkb, Wvb, bq, bk, bv, qscale, kscale, Qb, Kb, Vb);

  attn_kernel<<<dim3(T_ / 64, B_ * H_), 128, 0, stream>>>(Qb, Kb, Vb, mask, Xb);

  out_gemm<<<dim3((B_ * T_) / 128, F_ / 64), 128, 0, stream>>>(Xb, Wob, bo,
                                                               out);
}
